// VQVAE_15118284882613
// MI455X (gfx1250) — matrix operand order repaired and checked by reading, not compiled, not run on hardware
//
#include <hip/hip_runtime.h>

typedef __attribute__((ext_vector_type(16))) __bf16 v16bf;
typedef __attribute__((ext_vector_type(8)))  float  v8f;
typedef unsigned short u16;
typedef unsigned int   u32;

struct alignas(16) U4 { u32 a, b, c, d; };

__device__ __forceinline__ u16 f2bf(float f) {
  u32 u = __float_as_uint(f);
  u32 r = u + 0x7FFFu + ((u >> 16) & 1u);   // round-to-nearest-even
  return (u16)(r >> 16);
}
__device__ __forceinline__ float bf2f(u16 h) {
  return __uint_as_float(((u32)h) << 16);
}

// ---------------------------------------------------------------------------
// Implicit-GEMM conv kernel. NHWC bf16 activations; weights pre-packed in
// WMMA-fragment order (u32 = bf16 pair per lane/q). 256 threads = 8 waves,
// block tile 64(M) x 64(N), K-step 64 (4 WMMAs per barrier pair per wave).
// flags: 1 = ReLU, 2 = store f32 row-major [M][out_ld], 4 = store f32 NCHW.
// ---------------------------------------------------------------------------
struct CGArgs {
  const u16* A; const u32* Bp; const float* bias; const u16* resid;
  u16* obf; float* of32;
  int Nimg, IH, IW, in_ld;
  int KW, Cin, strd, padH, padW;
  int OH, OW, K, Kpad, Nout, Npack, nG, vec8;
  int out_step, ohoff, owoff, OHs, OWs;
  int out_ld, out_choff, res_ld, res_choff, flags;
};

__global__ __launch_bounds__(256) void conv_gemm_kernel(CGArgs a)
{
  __shared__ u16 As[64][66];       // [m][k], pitch 66 -> conflict-free frag reads
  __shared__ int KOFF[2048];       // k -> (kh*IW + kw)*in_ld + ci   (Kpad <= 2048)
  __shared__ int KHW[2048];        // k -> (kh<<16)|kw ; sentinel for k >= K

  const int tid  = threadIdx.x;
  const int lane = tid & 31, wave = tid >> 5;
  const int mw = wave & 3, nw = wave >> 2;          // 4x2 wave grid
  const int m0 = blockIdx.x * 64, n0 = blockIdx.y * 64;
  const int OHOW = a.OH * a.OW;
  const int M   = a.Nimg * OHOW;
  const int KWC = a.KW * a.Cin;

  // ---- one-time k-decode table (replaces per-iteration div/mod) ----
  for (int k = tid; k < a.Kpad; k += 256) {
    if (k < a.K) {
      int kh = k / KWC;  int r = k - kh * KWC;
      int kw = r / a.Cin; int ci = r - kw * a.Cin;
      KOFF[k] = (kh * a.IW + kw) * a.in_ld + ci;
      KHW[k]  = (kh << 16) | kw;
    } else { KOFF[k] = 0; KHW[k] = (0x7FFF << 16) | 0x7FFF; }
  }

  // ---- per-thread A staging assignment: one row, 16 consecutive k ----
  const int mA  = tid >> 2;
  const int kk0 = (tid & 3) * 16;
  int mg = m0 + mA;
  int nimg = 0, oh = 0, ow = 0;
  const bool mok = mg < M;
  if (mok) {
    nimg = mg / OHOW; int r = mg - nimg * OHOW;
    oh = r / a.OW; ow = r - oh * a.OW;
  }
  const int ihb = oh * a.strd - a.padH;
  const int iwb = ow * a.strd - a.padW;
  const long rowBaseE = (((long)nimg * a.IH + ihb) * a.IW + iwb) * a.in_ld;

  __syncthreads();   // KOFF/KHW ready

  v8f acc0 = {0.f,0.f,0.f,0.f,0.f,0.f,0.f,0.f};
  v8f acc1 = {0.f,0.f,0.f,0.f,0.f,0.f,0.f,0.f};

  const int half = lane >> 4, l16 = lane & 15;
  const int ng0 = (n0 >> 4) + nw * 2;   // 16-col group indices for this wave
  const int ng1 = ng0 + 1;

  for (int k0 = 0; k0 < a.Kpad; k0 += 64) {
    // ---- stage A tile (implicit im2col) ----
    if (a.vec8) {                         // Cin % 8 == 0: 8 k's share one tap
      #pragma unroll
      for (int g = 0; g < 16; g += 8) {
        int k = k0 + kk0 + g;
        int hw = KHW[k];
        int ih = ihb + (hw >> 16), iw = iwb + (hw & 0xFFFF);
        U4 val = {0u, 0u, 0u, 0u};
        if (mok && (unsigned)ih < (unsigned)a.IH && (unsigned)iw < (unsigned)a.IW)
          val = *(const U4*)(a.A + rowBaseE + KOFF[k]);
        u32* dstp = (u32*)&As[mA][kk0 + g];
        dstp[0] = val.a; dstp[1] = val.b; dstp[2] = val.c; dstp[3] = val.d;
      }
    } else {                              // stem layer (Cin = 3)
      #pragma unroll
      for (int j = 0; j < 16; ++j) {
        int k = k0 + kk0 + j;
        int hw = KHW[k];
        int ih = ihb + (hw >> 16), iw = iwb + (hw & 0xFFFF);
        u16 v = 0;
        if (mok && (unsigned)ih < (unsigned)a.IH && (unsigned)iw < (unsigned)a.IW)
          v = a.A[rowBaseE + KOFF[k]];
        As[mA][kk0 + j] = v;
      }
    }
    __syncthreads();

    // ---- two 32-K sub-steps: A frags from LDS, B frags direct from global ----
    #pragma unroll
    for (int sb = 0; sb < 2; ++sb) {
      const int kblk = (k0 >> 5) + sb;
      const int ksub = sb * 32;
      union Frag { U4 x[2]; u32 u[8]; v16bf v; };
      Frag af, b0, b1;
      const int arow = mw * 16 + l16;
      #pragma unroll
      for (int q = 0; q < 8; ++q) {
        int kk = ksub + half * 16 + q * 2;
        af.u[q] = *(const u32*)&As[arow][kk];
      }
      const u32* bp0 = a.Bp + (((size_t)kblk * a.nG + ng0) * 32 + lane) * 8;
      const u32* bp1 = a.Bp + (((size_t)kblk * a.nG + ng1) * 32 + lane) * 8;
      b0.x[0] = ((const U4*)bp0)[0]; b0.x[1] = ((const U4*)bp0)[1];
      b1.x[0] = ((const U4*)bp1)[0]; b1.x[1] = ((const U4*)bp1)[1];
      __builtin_prefetch(bp0 + (size_t)a.nG * 256, 0, 1);  // next k-block
      acc0 = __builtin_amdgcn_wmma_f32_16x16x32_bf16(false, af.v, false, b0.v,
                                                     (short)0, acc0, false, false);
      acc1 = __builtin_amdgcn_wmma_f32_16x16x32_bf16(false, af.v, false, b1.v,
                                                     (short)0, acc1, false, false);
    }
    __syncthreads();
  }

  // ---- epilogue: bias + residual + ReLU + store (incremental pixel decode) ----
  int base = m0 + mw * 16 + half * 8;
  if (base < M) {
    int ni = base / OHOW; int r = base - ni * OHOW;
    int oh2 = r / a.OW, ow2 = r - oh2 * a.OW;
    const int gn0 = n0 + nw * 32 + l16, gn1 = gn0 + 16;
    const float bv0 = (a.bias && gn0 < a.Nout) ? a.bias[gn0] : 0.f;
    const float bv1 = (a.bias && gn1 < a.Nout) ? a.bias[gn1] : 0.f;
    #pragma unroll
    for (int q = 0; q < 8; ++q) {
      int mg2 = base + q;
      if (mg2 < M) {
        int ohs = oh2 * a.out_step + a.ohoff;
        int ows = ow2 * a.out_step + a.owoff;
        size_t pix = ((size_t)ni * a.OHs + ohs) * a.OWs + ows;
        #pragma unroll
        for (int t = 0; t < 2; ++t) {
          int ng = t ? gn1 : gn0;
          if (ng < a.Nout) {
            float val = (t ? acc1[q] : acc0[q]) + (t ? bv1 : bv0);
            if (a.resid) val += bf2f(a.resid[pix * a.res_ld + a.res_choff + ng]);
            if (a.flags & 1) val = val > 0.f ? val : 0.f;
            if (a.flags & 2)      a.of32[(size_t)mg2 * a.out_ld + ng] = val;
            else if (a.flags & 4) a.of32[(((size_t)ni * a.Nout + ng) * a.OHs + ohs) * a.OWs + ows] = val;
            else                  a.obf[pix * a.out_ld + a.out_choff + ng] = f2bf(val);
          }
        }
      }
      if (++ow2 == a.OW) { ow2 = 0; if (++oh2 == a.OH) { oh2 = 0; ++ni; } }
    }
  }
}

// ---------------------------------------------------------------------------
// Weight repack: fp32 OIHW -> bf16 pairs in WMMA-fragment order:
//   dst[((kblk*nG + ng)*32 + lane)*8 + q] = { k = kblk*32 + (lane/16)*16 + 2q,
//                                             n = ng*16 + lane%16 } pair.
// Sub-kernel selection (khoff/step etc.) implements convT parity classes.
// ---------------------------------------------------------------------------
__global__ void pack_weight_kernel(const float* __restrict__ src, u32* __restrict__ dst,
                                   int O, int I, int KH, int KW, int KHs, int KWs,
                                   int khoff, int khstep, int kwoff, int kwstep,
                                   int K, int Kpad, int Npack)
{
  int nG = Npack >> 4;
  long total = (long)(Kpad >> 5) * nG * 256;
  long idx = (long)blockIdx.x * blockDim.x + threadIdx.x;
  if (idx >= total) return;
  int q    = (int)(idx & 7);
  int lane = (int)((idx >> 3) & 31);
  long t   = idx >> 8;
  int ng   = (int)(t % nG);
  int kblk = (int)(t / nG);
  int n     = ng * 16 + (lane & 15);
  int kbase = kblk * 32 + (lane >> 4) * 16 + q * 2;
  u32 out = 0;
  #pragma unroll
  for (int e = 0; e < 2; ++e) {
    int k = kbase + e;
    u16 v = 0;
    if (k < K && n < O) {
      int kh_i = k / (KWs * I); int r = k - kh_i * (KWs * I);
      int kw_i = r / I;         int ci = r - kw_i * I;
      int kh = khoff + kh_i * khstep, kw = kwoff + kw_i * kwstep;
      v = f2bf(src[(((size_t)n * I + ci) * KH + kh) * KW + kw]);
    }
    out |= ((u32)v) << (16 * e);
  }
  dst[idx] = out;
}

__global__ void nchw_to_nhwc_kernel(const float* __restrict__ src, u16* __restrict__ dst,
                                    int N, int C, int H, int W)
{
  long total = (long)N * C * H * W;
  for (long idx = (long)blockIdx.x * blockDim.x + threadIdx.x; idx < total;
       idx += (long)gridDim.x * blockDim.x) {
    int c = (int)(idx % C); long r = idx / C;
    int w = (int)(r % W); r /= W;
    int h = (int)(r % H); int n = (int)(r / H);
    dst[idx] = f2bf(src[(((size_t)n * C + c) * H + h) * W + w]);
  }
}

__global__ void copy_ch_kernel(const u16* __restrict__ src, int src_ld, int src_off,
                               u16* __restrict__ dst, int dst_ld, int dst_off,
                               int C, long P)
{
  long total = P * C;
  for (long idx = (long)blockIdx.x * blockDim.x + threadIdx.x; idx < total;
       idx += (long)gridDim.x * blockDim.x) {
    long p = idx / C; int c = (int)(idx - p * C);
    dst[p * dst_ld + dst_off + c] = src[p * src_ld + src_off + c];
  }
}

__global__ void cbnorm_kernel(const float* __restrict__ cb, float* __restrict__ out,
                              int Kcb, int D)
{
  int k = blockIdx.x * blockDim.x + threadIdx.x;
  if (k >= Kcb) return;
  float s = 0.f;
  for (int d = 0; d < D; ++d) { float v = cb[(size_t)k * D + d]; s += v * v; }
  out[k] = s;
}

__global__ void zero2_kernel(float* p) { if (threadIdx.x < 2) p[threadIdx.x] = 0.f; }

// One wave32 per token: argmin over codebook (wnorm - 2*dot), gather zq,
// accumulate commitment-loss partials.
__global__ __launch_bounds__(256) void vq_assign_kernel(
    const float* __restrict__ scores, const u16* __restrict__ ze,
    const float* __restrict__ cb, const float* __restrict__ wnorm,
    u16* __restrict__ outmap, int out_ld, int out_choff,
    float* __restrict__ loss_acc, int M, int Kcb, int D)
{
  int wave = threadIdx.x >> 5, lane = threadIdx.x & 31;
  int m = blockIdx.x * 8 + wave;
  if (m >= M) return;
  float bestv = 3.4e38f; int besti = 0x7FFFFFFF;
  for (int c = lane; c < Kcb; c += 32) {
    float d = wnorm[c] - 2.f * scores[(size_t)m * Kcb + c];
    if (d < bestv || (d == bestv && c < besti)) { bestv = d; besti = c; }
  }
  for (int off = 16; off; off >>= 1) {
    float ov = __shfl_xor(bestv, off, 32);
    int   oi = __shfl_xor(besti, off, 32);
    if (ov < bestv || (ov == bestv && oi < besti)) { bestv = ov; besti = oi; }
  }
  int idx = __shfl(besti, 0, 32);
  float lsum = 0.f;
  for (int d = lane; d < D; d += 32) {
    float zq  = cb[(size_t)idx * D + d];
    float zev = bf2f(ze[(size_t)m * D + d]);
    outmap[(size_t)m * out_ld + out_choff + d] = f2bf(zq);
    float df = zev - zq; lsum += df * df;
  }
  for (int off = 16; off; off >>= 1) lsum += __shfl_xor(lsum, off, 32);
  if (lane == 0) atomicAdd(loss_acc, lsum);
}

__global__ void finalize_kernel(const float* loss, float* out)
{
  out[0] = loss[0] * (1.f / 524288.f) + loss[1] * (1.f / 2097152.f);
}

// ---------------------------------------------------------------------------
extern "C" void kernel_launch(void* const* d_in, const int* in_sizes, int n_in,
                              void* d_out, int out_size, void* d_ws, size_t ws_size,
                              hipStream_t stream)
{
  (void)in_sizes; (void)n_in; (void)ws_size;
  hipStream_t s = stream;
  auto Wf = [&](int i) { return (const float*)d_in[i]; };
  const float* x = Wf(0);

  // ---- workspace carving (deterministic per call) ----
  size_t off = 0; char* base = (char*)d_ws;
  auto alloc = [&](size_t bytes) -> void* {
    void* p = base + off; off += (bytes + 255) & ~(size_t)255; return p;
  };
  auto abf = [&](size_t elems) -> u16* { return (u16*)alloc(elems * sizeof(u16)); };

  u16* ACT  = abf(1572864);   // x NHWC           [8,256,256,3]
  u16* A1   = abf(8388608);   //                  [8,128,128,64]
  u16* A2   = abf(4194304);   //                  [8,64,64,128]
  u16* A3   = abf(4194304);
  u16* A4   = abf(4194304);
  u16* MID  = abf(2097152);   // resblock hidden  [8,64,64,64] max
  u16* T1   = abf(524288);    //                  [8,32,32,64]
  u16* T2   = abf(1048576);   //                  [8,32,32,128]
  u16* T3   = abf(1048576);
  u16* TQ   = abf(524288);    // top tokens       [8192,64]
  u16* TQM  = abf(524288);    // quantized top map
  u16* TD1  = abf(1048576);
  u16* TD2  = abf(1048576);
  u16* CAT  = abf(6291456);   //                  [8,64,64,192]
  u16* BQ   = abf(2097152);   // bottom tokens    [32768,64]
  u16* CAT2 = abf(4194304);   //                  [8,64,64,128]
  u16* BD1  = abf(4194304);
  u16* BD2  = abf(4194304);
  u16* U1   = abf(8388608);   //                  [8,128,128,64]
  float* SC   = (float*)alloc((size_t)32768 * 512 * 4);
  float* WN   = (float*)alloc(512 * 4);
  float* LOSS = (float*)alloc(2 * 4);

  auto pack = [&](const float* src, int O, int I, int KH, int KW,
                  int KHs, int KWs, int khoff, int khstep, int kwoff, int kwstep) -> const u32* {
    int K = KHs * KWs * I, Kpad = (K + 63) & ~63, Npack = (O + 63) & ~63;
    u32* dst = (u32*)alloc((size_t)Kpad * Npack * 2);
    long total = (long)(Kpad >> 5) * (Npack >> 4) * 256;
    pack_weight_kernel<<<(int)((total + 255) / 256), 256, 0, s>>>(
        src, dst, O, I, KH, KW, KHs, KWs, khoff, khstep, kwoff, kwstep, K, Kpad, Npack);
    return dst;
  };

  auto gemm = [&](const u16* A, const u32* Bw, const float* bias, const u16* resid,
                  u16* obf, float* of32,
                  int Nimg, int IH, int IW, int in_ld,
                  int KW, int Cin, int strd, int padH, int padW,
                  int OH, int OW, int K, int Nout,
                  int out_step, int ohoff, int owoff, int OHs, int OWs,
                  int out_ld, int out_choff, int res_ld, int res_choff, int flags) {
    int Kpad = (K + 63) & ~63, Npack = (Nout + 63) & ~63;
    CGArgs a{A, Bw, bias, resid, obf, of32, Nimg, IH, IW, in_ld,
             KW, Cin, strd, padH, padW, OH, OW, K, Kpad, Nout, Npack, Npack >> 4,
             (Cin % 8) == 0 ? 1 : 0,
             out_step, ohoff, owoff, OHs, OWs, out_ld, out_choff, res_ld, res_choff, flags};
    int M = Nimg * OH * OW;
    dim3 g((M + 63) / 64, Npack / 64);
    conv_gemm_kernel<<<g, 256, 0, s>>>(a);
  };

  auto convstd = [&](const u16* A, const float* wsrc, const float* bias,
                     int O, int I, int KH, int KW, int strd, int pad,
                     int Nimg, int IH, int IW, int in_ld,
                     u16* out, int out_ld, int out_choff, int flags,
                     const u16* resid, int res_ld, int res_choff) {
    const u32* Bw = pack(wsrc, O, I, KH, KW, KH, KW, 0, 1, 0, 1);
    int OH = (IH + 2 * pad - KH) / strd + 1;
    int OW = (IW + 2 * pad - KW) / strd + 1;
    gemm(A, Bw, bias, resid, out, nullptr, Nimg, IH, IW, in_ld, KW, I, strd, pad, pad,
         OH, OW, KH * KW * I, O, 1, 0, 0, OH, OW, out_ld, out_choff, res_ld, res_choff, flags);
  };

  // ConvTranspose2d(k=4,s=2,p=1) as 4 parity-class 2x2 convs.
  auto convT4 = [&](const u16* A, const float* wsrc, const float* bias, int O, int I,
                    int Nimg, int IH, int IW, int in_ld,
                    u16* out, int out_ld, int out_choff, int flags, float* of32) {
    for (int aa = 0; aa < 2; ++aa)
      for (int bb = 0; bb < 2; ++bb) {
        const u32* Bw = pack(wsrc, O, I, 4, 4, 2, 2, aa, 2, bb, 2);
        gemm(A, Bw, bias, nullptr, out, of32, Nimg, IH, IW, in_ld,
             2, I, 1, 1 - aa, 1 - bb, IH, IW, 4 * I, O,
             2, aa, bb, 2 * IH, 2 * IW, out_ld, out_choff, 0, 0, flags);
      }
  };

  // resblock: h = relu(conv3x3 128->64); out = relu(conv1x1 64->128 + x)
  auto resblock = [&](const u16* xin, u16* out,
                      const float* w1, const float* b1, const float* w2, const float* b2,
                      int Nimg, int H, int Wd) {
    convstd(xin, w1, b1, 64, 128, 3, 3, 1, 1, Nimg, H, Wd, 128, MID, 64, 0, 1, nullptr, 0, 0);
    convstd(MID, w2, b2, 128, 64, 1, 1, 1, 0, Nimg, H, Wd, 64, out, 128, 0, 1, xin, 128, 0);
  };

  // ---- forward pass ----
  nchw_to_nhwc_kernel<<<4096, 256, 0, s>>>(x, ACT, 8, 3, 256, 256);
  zero2_kernel<<<1, 32, 0, s>>>(LOSS);

  // bottom encoder
  convstd(ACT, Wf(1),  Wf(2),  64,   3, 4, 4, 2, 1, 8, 256, 256,   3, A1,  64, 0, 1, nullptr, 0, 0);
  convstd(A1,  Wf(3),  Wf(4),  128, 64, 4, 4, 2, 1, 8, 128, 128,  64, A2, 128, 0, 1, nullptr, 0, 0);
  convstd(A2,  Wf(5),  Wf(6),  128,128, 3, 3, 1, 1, 8,  64,  64, 128, A3, 128, 0, 1, nullptr, 0, 0);
  resblock(A3, A4, Wf(7),  Wf(8),  Wf(9),  Wf(10), 8, 64, 64);
  resblock(A4, A3, Wf(11), Wf(12), Wf(13), Wf(14), 8, 64, 64);   // b_enc = A3

  // top encoder
  convstd(A3, Wf(15), Wf(16), 64, 128, 4, 4, 2, 1, 8, 64, 64, 128, T1,  64, 0, 1, nullptr, 0, 0);
  convstd(T1, Wf(17), Wf(18), 128, 64, 3, 3, 1, 1, 8, 32, 32,  64, T2, 128, 0, 1, nullptr, 0, 0);
  resblock(T2, T3, Wf(19), Wf(20), Wf(21), Wf(22), 8, 32, 32);
  resblock(T3, T2, Wf(23), Wf(24), Wf(25), Wf(26), 8, 32, 32);
  convstd(T2, Wf(27), Wf(28), 64, 128, 1, 1, 1, 0, 8, 32, 32, 128, TQ, 64, 0, 0, nullptr, 0, 0);

  // quantize top
  cbnorm_kernel<<<8, 64, 0, s>>>(Wf(29), WN, 512, 64);
  {
    const u32* CBt = pack(Wf(29), 512, 64, 1, 1, 1, 1, 0, 1, 0, 1);
    gemm(TQ, CBt, nullptr, nullptr, nullptr, SC, 8, 32, 32, 64, 1, 64, 1, 0, 0,
         32, 32, 64, 512, 1, 0, 0, 32, 32, 512, 0, 0, 0, 2);
  }
  vq_assign_kernel<<<(8192 + 7) / 8, 256, 0, s>>>(SC, TQ, Wf(29), WN, TQM, 64, 0,
                                                  LOSS + 0, 8192, 512, 64);

  // top decoder -> CAT[:,0:64)
  convstd(TQM, Wf(30), Wf(31), 128, 64, 3, 3, 1, 1, 8, 32, 32, 64, TD1, 128, 0, 1, nullptr, 0, 0);
  resblock(TD1, TD2, Wf(32), Wf(33), Wf(34), Wf(35), 8, 32, 32);
  resblock(TD2, TD1, Wf(36), Wf(37), Wf(38), Wf(39), 8, 32, 32);
  convT4(TD1, Wf(40), Wf(41), 64, 128, 8, 32, 32, 128, CAT, 192, 0, 0, nullptr);

  // concat b_enc -> CAT[:,64:192)
  copy_ch_kernel<<<4096, 256, 0, s>>>(A3, 128, 0, CAT, 192, 64, 128, (long)8 * 64 * 64);

  // bottom quantizer conv + quantize (b_q -> CAT2[:,64:128))
  convstd(CAT, Wf(42), Wf(43), 64, 192, 1, 1, 1, 0, 8, 64, 64, 192, BQ, 64, 0, 0, nullptr, 0, 0);
  cbnorm_kernel<<<8, 64, 0, s>>>(Wf(44), WN, 512, 64);
  {
    const u32* CBb = pack(Wf(44), 512, 64, 1, 1, 1, 1, 0, 1, 0, 1);
    gemm(BQ, CBb, nullptr, nullptr, nullptr, SC, 8, 64, 64, 64, 1, 64, 1, 0, 0,
         64, 64, 64, 512, 1, 0, 0, 64, 64, 512, 0, 0, 0, 2);
  }
  vq_assign_kernel<<<(32768 + 7) / 8, 256, 0, s>>>(SC, BQ, Wf(44), WN, CAT2, 128, 64,
                                                   LOSS + 1, 32768, 512, 64);

  // upsample t_q -> CAT2[:,0:64)
  convT4(TQM, Wf(45), Wf(46), 64, 64, 8, 32, 32, 64, CAT2, 128, 0, 0, nullptr);

  // bottom decoder
  convstd(CAT2, Wf(47), Wf(48), 128, 128, 3, 3, 1, 1, 8, 64, 64, 128, BD1, 128, 0, 1, nullptr, 0, 0);
  resblock(BD1, BD2, Wf(49), Wf(50), Wf(51), Wf(52), 8, 64, 64);
  resblock(BD2, BD1, Wf(53), Wf(54), Wf(55), Wf(56), 8, 64, 64);
  convT4(BD1, Wf(57), Wf(58), 64, 128, 8, 64, 64, 128, U1, 64, 0, 1, nullptr);
  // final convT: 64->3, ReLU, store f32 NCHW directly to d_out
  convT4(U1, Wf(59), Wf(60), 3, 64, 8, 128, 128, 64, nullptr, 0, 0, 1 | 4, (float*)d_out);

  // dists scalar
  finalize_kernel<<<1, 1, 0, s>>>(LOSS, (float*)d_out + (out_size - 1));
}
